// ReccurentDecoder_81381040324959
// MI455X (gfx1250) — matrix operand order repaired and checked by reading, not compiled, not run on hardware
//
#include <hip/hip_runtime.h>
#include <hip/hip_bf16.h>

// Problem constants
#define Bsz   256
#define Wlen  64
#define Tlen  512
#define Dd    256
#define Hh    512
#define MHd   512
#define DKP   288   // D+O=257 padded to multiple of 32
#define XUS   288   // xu row stride (halves)
#define NSTEPS 575  // 64 warmup + 511 autoregressive

typedef __attribute__((ext_vector_type(16))) __bf16 v16bf;
typedef __attribute__((ext_vector_type(8)))  float  v8f;

__device__ __forceinline__ unsigned short f2bf(float f) {
  unsigned int u = __float_as_uint(f);
  if ((u & 0x7fffffffu) > 0x7f800000u) return (unsigned short)0x7fc0; // NaN
  unsigned int r = (u + 0x7fffu + ((u >> 16) & 1u)) >> 16;            // RNE
  return (unsigned short)r;
}

// Opaque zero (SGPR). Adding it to a weight pointer inside the step loop
// prevents LICM from hoisting the weight loads out of the 575-iteration loop
// (which previously spilled hundreds of fragments to scratch), while keeping
// the pointer's global address-space provenance so loads lower to
// global_load_b128 (LOADcnt only) instead of flat_load_b128 (LOADcnt+DScnt,
// which falsely orders the weight stream against the LDS A-fragment stream).
__device__ __forceinline__ unsigned int opq0() {
  unsigned int v = 0;
  asm volatile("" : "+s"(v));
  return v;
}

union FragU { uint4 q[2]; v16bf v; };

// A-fragment (16x32 bf16, M=16,K=32) from LDS row-major [row][k], stride in halves.
// Lanes 0-15 hold K 0-15 contiguous; lanes 16-31: K 16-31 (matches B fragment order).
__device__ __forceinline__ v16bf frag_a_lds(const unsigned short* base, int stride,
                                            int lane, int k0) {
  const unsigned short* p = base + (lane & 15) * stride + k0 + (lane >> 4) * 16;
  FragU f;
  f.q[0] = *(const uint4*)(p);
  f.q[1] = *(const uint4*)(p + 8);
  return f.v;
}

// B-fragment (32x16 bf16, K=32,N=16) from global weights stored row-major Wt[n][k]
// (B[k][n] = Wt[n][k]). Lane n holds column n: 16 contiguous K halves
// (lanes 0-15: K k0..k0+15, lanes 16-31: K k0+16..k0+31).
__device__ __forceinline__ v16bf frag_b_glb(const unsigned short* wt, int stride,
                                            int lane, int n0, int k0) {
  const unsigned short* p = wt + (size_t)(n0 + (lane & 15)) * (size_t)stride
                               + k0 + (lane >> 4) * 16;
  FragU f;
  f.q[0] = *(const uint4*)(p);
  f.q[1] = *(const uint4*)(p + 8);
  return f.v;
}

__device__ __forceinline__ v8f wmma_acc(v8f c, v16bf a, v16bf b) {
  return __builtin_amdgcn_wmma_f32_16x16x32_bf16(
      /*neg_a=*/false, a, /*neg_b=*/false, b,
      /*c_mod=*/(short)0, c, /*reuse_a=*/false, /*reuse_b=*/false);
}

// One K-sweep of a 16x16 output tile: A from LDS, B rows [n0, n0+16) of weights.
__device__ __forceinline__ v8f gemm_rows(const unsigned short* aBase, int aStride, int aK,
                                         const unsigned short* w, int wStride,
                                         int n0, int lane, v8f acc) {
  for (int k0 = 0; k0 < aK; k0 += 32)
    acc = wmma_acc(acc, frag_a_lds(aBase, aStride, lane, k0),
                        frag_b_glb(w, wStride, lane, n0, k0));
  return acc;
}

__device__ __forceinline__ float sigf(float x) { return 1.0f / (1.0f + __expf(-x)); }
__device__ __forceinline__ float eluf(float x) { return x > 0.0f ? x : (__expf(x) - 1.0f); }

// One-time fp32 -> bf16 weight conversion with zero K-padding.
__global__ void cvt_pad_bf16(const float* __restrict__ src, unsigned short* __restrict__ dst,
                             int rows, int cols, int dcols) {
  int total = rows * dcols;
  for (int i = blockIdx.x * blockDim.x + threadIdx.x; i < total;
       i += gridDim.x * blockDim.x) {
    int r = i / dcols, c = i - r * dcols;
    dst[i] = (c < cols) ? f2bf(src[r * cols + c]) : (unsigned short)0;
  }
}

// Persistent kernel: one workgroup per 16-row batch tile, 16 waves, each wave
// owns a 32-wide hidden-column group (two 16-col subtiles). Runs all 575
// sequential steps with h resident in LDS; weights stream from L2 every step.
__global__ __launch_bounds__(512, 1)
void gru_decoder_kernel(const float* __restrict__ x,
                        const float* __restrict__ xw,
                        const float* __restrict__ yw,
                        const unsigned short* __restrict__ wih_,  // [1536][288] bf16
                        const unsigned short* __restrict__ whh_,  // [1536][512] bf16
                        const float* __restrict__ b_ih,
                        const float* __restrict__ b_hh,
                        const unsigned short* __restrict__ w1_,   // [512][512] bf16
                        const float* __restrict__ b1,
                        const unsigned short* __restrict__ w2_,   // [512][512] bf16
                        const float* __restrict__ b2,
                        const float* __restrict__ wf,             // [512] fp32
                        const float* __restrict__ bfv,
                        float* __restrict__ out)                  // [256][512] fp32
{
  __shared__ float          h_f[16 * Hh];   // fp32 master hidden state
  __shared__ unsigned short h_b[16 * Hh];   // bf16 mirror (WMMA A source)
  __shared__ unsigned short xu [16 * XUS];  // step input [x_t | prev | pad], bf16
  __shared__ unsigned short a1b[16 * MHd];  // MLP layer-1 activations, bf16
  __shared__ float          a2f[16 * MHd];  // MLP layer-2 activations, fp32
  __shared__ float          predv[16];

  const int tid  = threadIdx.x;
  const int lane = tid & 31;
  const int wave = tid >> 5;         // 0..15
  const int wg   = blockIdx.x;       // 0..15 (batch tile)
  const int c0   = wave * 32;        // this wave's 32-wide hidden-column group

  // init: h = 0, xu pad = 0
  for (int i = tid; i < 16 * Hh; i += 512) { h_f[i] = 0.0f; h_b[i] = 0; }
  for (int i = tid; i < 16 * XUS; i += 512) xu[i] = 0;
  if (tid < 16) predv[tid] = 0.0f;
  __syncthreads();

  // step-0 input: x_warmup[:,1,:], y = y_warmup[:,0]
  {
    int row = tid >> 5, kk = (tid & 31) * 8;   // 16 rows x 32 threads x 8 floats
    int b = wg * 16 + row;
    const float* src = &xw[((size_t)b * Wlen + 1) * Dd + kk];
    float4 v0 = *(const float4*)(src);
    float4 v1 = *(const float4*)(src + 4);
    unsigned short* dst = &xu[row * XUS + kk];
    dst[0] = f2bf(v0.x); dst[1] = f2bf(v0.y); dst[2] = f2bf(v0.z); dst[3] = f2bf(v0.w);
    dst[4] = f2bf(v1.x); dst[5] = f2bf(v1.y); dst[6] = f2bf(v1.z); dst[7] = f2bf(v1.w);
    if (tid < 16)
      xu[tid * XUS + 256] = f2bf(yw[(size_t)(wg * 16 + tid) * Wlen + 0]);
  }
  __syncthreads();

  const int mbase = (lane >> 4) * 8;   // C-layout: VGPR k -> row mbase+k

  for (int s = 0; s < NSTEPS; ++s) {
    // Per-step opaque offset: keeps weight loads inside the loop (L2 hits),
    // as global_load_b128 (provenance preserved).
    const unsigned short* wih = wih_ + opq0();
    const unsigned short* whh = whh_ + opq0();

    // ---- GRU gates: two 16-col subtiles ----
    float hnew[2][8];
#pragma unroll
    for (int sub = 0; sub < 2; ++sub) {
      const int n0  = c0 + sub * 16;
      const int col = n0 + (lane & 15);

      // r gate: gi + gh fused accumulation
      v8f acc = {};
      acc = gemm_rows(h_b, Hh, Hh,  whh, Hh,  n0, lane, acc);
      acc = gemm_rows(xu, XUS, DKP, wih, DKP, n0, lane, acc);
      float rg[8];
      {
        float bias = b_ih[col] + b_hh[col];
#pragma unroll
        for (int k = 0; k < 8; ++k) rg[k] = sigf(acc[k] + bias);
      }
      // z gate
      acc = (v8f){};
      acc = gemm_rows(h_b, Hh, Hh,  whh, Hh,  Hh + n0, lane, acc);
      acc = gemm_rows(xu, XUS, DKP, wih, DKP, Hh + n0, lane, acc);
      float zg[8];
      {
        float bias = b_ih[Hh + col] + b_hh[Hh + col];
#pragma unroll
        for (int k = 0; k < 8; ++k) zg[k] = sigf(acc[k] + bias);
      }
      // n gate: h-part
      acc = (v8f){};
      acc = gemm_rows(h_b, Hh, Hh, whh, Hh, 2 * Hh + n0, lane, acc);
      float nh[8];
      {
        float bias = b_hh[2 * Hh + col];
#pragma unroll
        for (int k = 0; k < 8; ++k) nh[k] = acc[k] + bias;
      }
      // n gate: x-part + combine
      acc = (v8f){};
      acc = gemm_rows(xu, XUS, DKP, wih, DKP, 2 * Hh + n0, lane, acc);
      {
        float bias = b_ih[2 * Hh + col];
#pragma unroll
        for (int k = 0; k < 8; ++k) {
          float n  = tanhf(acc[k] + bias + rg[k] * nh[k]);
          float ho = h_f[(mbase + k) * Hh + col];
          hnew[sub][k] = (1.0f - zg[k]) * n + zg[k] * ho;
        }
      }
    }
    __syncthreads();   // all reads of h_b/h_f/xu for this step are done
#pragma unroll
    for (int sub = 0; sub < 2; ++sub) {
      const int col = c0 + sub * 16 + (lane & 15);
#pragma unroll
      for (int k = 0; k < 8; ++k) {
        h_f[(mbase + k) * Hh + col] = hnew[sub][k];
        h_b[(mbase + k) * Hh + col] = f2bf(hnew[sub][k]);
      }
    }

    const bool do_mlp = (s >= 63);   // reference only emits MLP from step 63 onward
    if (do_mlp) {
      const unsigned short* w1 = w1_ + opq0();
      const unsigned short* w2 = w2_ + opq0();
      __syncthreads();               // new h_b visible
      // MLP layer 1: elu(h @ W1^T + b1)
#pragma unroll
      for (int sub = 0; sub < 2; ++sub) {
        const int n0  = c0 + sub * 16;
        const int col = n0 + (lane & 15);
        v8f acc = {};
        acc = gemm_rows(h_b, Hh, Hh, w1, Hh, n0, lane, acc);
        float bias = b1[col];
#pragma unroll
        for (int k = 0; k < 8; ++k)
          a1b[(mbase + k) * MHd + col] = f2bf(eluf(acc[k] + bias));
      }
      __syncthreads();
      // MLP layer 2: elu(a1 @ W2^T + b2)
#pragma unroll
      for (int sub = 0; sub < 2; ++sub) {
        const int n0  = c0 + sub * 16;
        const int col = n0 + (lane & 15);
        v8f acc = {};
        acc = gemm_rows(a1b, MHd, MHd, w2, MHd, n0, lane, acc);
        float bias = b2[col];
#pragma unroll
        for (int k = 0; k < 8; ++k)
          a2f[(mbase + k) * MHd + col] = eluf(acc[k] + bias);
      }
      __syncthreads();
      // final projection: a2 @ Wf^T + bf  (O=1 -> wave-level dot, wave = batch row)
      {
        float part = 0.0f;
        for (int k = lane; k < MHd; k += 32)
          part += a2f[wave * MHd + k] * wf[k];
#pragma unroll
        for (int off = 16; off > 0; off >>= 1)
          part += __shfl_xor(part, off, 32);
        if (lane == 0) {
          float pv = part + bfv[0];
          predv[wave] = pv;
          int tout = (s == 63) ? 0 : (s - 63);
          out[(size_t)(wg * 16 + wave) * Tlen + tout] = pv;
        }
      }
      __syncthreads();               // predv ready for feedback
    }

    // ---- prepare next step's input ----
    int ns = s + 1;
    if (ns < NSTEPS) {
      int row = tid >> 5, kk = (tid & 31) * 8;
      int b = wg * 16 + row;
      const float* src;
      if (ns < 63)       src = &xw[((size_t)b * Wlen + (ns + 1)) * Dd + kk]; // x_warmup[:,ns+1]
      else if (ns == 63) src = &x [((size_t)b * Tlen + 0) * Dd + kk];        // x[:,0]
      else               src = &x [((size_t)b * Tlen + (ns - 63)) * Dd + kk];// x[:,ns-63]
      float4 v0 = *(const float4*)(src);
      float4 v1 = *(const float4*)(src + 4);
      unsigned short* dst = &xu[row * XUS + kk];
      dst[0] = f2bf(v0.x); dst[1] = f2bf(v0.y); dst[2] = f2bf(v0.z); dst[3] = f2bf(v0.w);
      dst[4] = f2bf(v1.x); dst[5] = f2bf(v1.y); dst[6] = f2bf(v1.z); dst[7] = f2bf(v1.w);
      if (tid < 16) {
        int bb = wg * 16 + tid;
        float pv = (ns < 64) ? yw[(size_t)bb * Wlen + ns] : predv[tid];
        xu[tid * XUS + 256] = f2bf(pv);
      }
    }
    __syncthreads();
  }
}

extern "C" void kernel_launch(void* const* d_in, const int* in_sizes, int n_in,
                              void* d_out, int out_size, void* d_ws, size_t ws_size,
                              hipStream_t stream) {
  (void)in_sizes; (void)n_in; (void)out_size; (void)ws_size;
  const float* x   = (const float*)d_in[0];
  const float* xw  = (const float*)d_in[1];
  const float* yw  = (const float*)d_in[2];
  const float* Wih = (const float*)d_in[3];
  const float* Whh = (const float*)d_in[4];
  const float* bih = (const float*)d_in[5];
  const float* bhh = (const float*)d_in[6];
  const float* W1  = (const float*)d_in[7];
  const float* b1  = (const float*)d_in[8];
  const float* W2  = (const float*)d_in[9];
  const float* b2  = (const float*)d_in[10];
  const float* Wf  = (const float*)d_in[11];
  const float* bfv = (const float*)d_in[12];
  float* out = (float*)d_out;

  unsigned char* ws = (unsigned char*)d_ws;
  unsigned short* wih_b = (unsigned short*)(ws + 0);        // 1536*288*2 = 884736 B
  unsigned short* whh_b = (unsigned short*)(ws + 884736);   // 1536*512*2 = 1572864 B
  unsigned short* w1_b  = (unsigned short*)(ws + 2457600);  // 512*512*2  = 524288 B
  unsigned short* w2_b  = (unsigned short*)(ws + 2981888);  // 512*512*2  = 524288 B

  cvt_pad_bf16<<<1728, 256, 0, stream>>>(Wih, wih_b, 1536, 257, DKP);
  cvt_pad_bf16<<<3072, 256, 0, stream>>>(Whh, whh_b, 1536, Hh, Hh);
  cvt_pad_bf16<<<1024, 256, 0, stream>>>(W1,  w1_b,  MHd, Hh, Hh);
  cvt_pad_bf16<<<1024, 256, 0, stream>>>(W2,  w2_b,  MHd, MHd, MHd);

  gru_decoder_kernel<<<16, 512, 0, stream>>>(
      x, xw, yw, wih_b, whh_b, bih, bhh, w1_b, b1, w2_b, b2, Wf, bfv, out);
}
